// FP8Experts_51943334478419
// MI455X (gfx1250) — 8-bit matrix operand order repaired and checked by reading, compiled, not run on hardware
//
#include <hip/hip_runtime.h>
#include <hip/hip_bf16.h>

typedef __attribute__((ext_vector_type(16))) int   v16i;
typedef __attribute__((ext_vector_type(8)))  float v8f;

#define T_DIM 2048
#define H_DIM 2048
#define I_DIM 1408
#define NKB_H 16
#define NKB_I 11


__device__ __forceinline__ float warp_max32(float v) {
#pragma unroll
  for (int off = 16; off >= 1; off >>= 1)
    v = fmaxf(v, __shfl_xor(v, off, 32));
  return v;
}

__device__ __forceinline__ unsigned int f32_to_e4m3_1(float x) {
  unsigned int s = (__float_as_uint(x) >> 24) & 0x80u;
  float ax = fminf(fabsf(x), 448.f);
  if (!(ax >= 0x1p-10f)) return s;
  int e;
  float m = frexpf(ax, &e);
  int E = e + 6;
  unsigned int bits;
  if (E <= 0) {
    int f = (int)rintf(ldexpf(ax, 9));
    bits = (f >= 8) ? 0x08u : (unsigned)f;
  } else {
    int f = (int)rintf(m * 16.f) - 8;
    if (f >= 8) { f = 0; E += 1; }
    if (E > 15) { E = 15; f = 6; }
    bits = ((unsigned)E << 3) | (unsigned)f;
  }
  return bits | s;
}

__device__ __forceinline__ unsigned int pack_fp8x4(float a, float b, float c, float d) {
#if __has_builtin(__builtin_amdgcn_cvt_pk_fp8_f32)
  int r = __builtin_amdgcn_cvt_pk_fp8_f32(a, b, 0, false);
  r = __builtin_amdgcn_cvt_pk_fp8_f32(c, d, r, true);
  return (unsigned)r;
#else
  return f32_to_e4m3_1(a) | (f32_to_e4m3_1(b) << 8) |
         (f32_to_e4m3_1(c) << 16) | (f32_to_e4m3_1(d) << 24);
#endif
}

__device__ __forceinline__ unsigned lds_off_of(const void* p) {
  return (unsigned)(unsigned long long)(__attribute__((address_space(3))) const char*)p;
}

__device__ __forceinline__ void async_copy_b128(const void* gsrc, unsigned ldsdst) {
  asm volatile("global_load_async_to_lds_b128 %0, %1, off"
               :: "v"(ldsdst), "v"(gsrc) : "memory");
}
__device__ __forceinline__ void wait_async_le0()  { asm volatile("s_wait_asynccnt 0x0" ::: "memory"); }
__device__ __forceinline__ void wait_async_le6()  { asm volatile("s_wait_asynccnt 0x6" ::: "memory"); }
__device__ __forceinline__ void wait_async_le10() { asm volatile("s_wait_asynccnt 0xa" ::: "memory"); }

__device__ __forceinline__ v16i lds_load_a(const unsigned char* tile, int lane) {
  const int r  = lane & 15;
  const int hi = (lane >> 4) << 3;
  v16i a;
  unsigned long long* au = reinterpret_cast<unsigned long long*>(&a);
#pragma unroll
  for (int j = 0; j < 8; ++j)
    au[j] = *reinterpret_cast<const unsigned long long*>(tile + r * 128 + j * 16 + hi);
  return a;
}

__device__ __forceinline__ v16i lds_load_b_tok(const unsigned char* xt, int lane, int mloc) {
  v16i b;
  unsigned char* bp = reinterpret_cast<unsigned char*>(&b);
  const int hi = (lane >> 4) << 3;
#pragma unroll
  for (int s = 0; s < 64; ++s)
    bp[s] = xt[(((s >> 3) << 4) + hi + (s & 7)) * 64 + mloc + (lane & 15)];
  return b;
}


__global__ void moe_zero(float* __restrict__ p, int n) {
  int i = blockIdx.x * 256 + threadIdx.x;
  if (i < n) p[i] = 0.f;
}

__global__ void moe_quant_x(const float* __restrict__ X,
                            unsigned char* __restrict__ XqT,
                            float* __restrict__ Xs) {
  const int lane = threadIdx.x & 31;
  const int w    = threadIdx.x >> 5;
  const int blk  = blockIdx.x * 8 + w;
  const int row  = blk >> 4;
  const int kb   = blk & 15;
  const float4 xv = *reinterpret_cast<const float4*>(X + row * H_DIM + kb * 128 + lane * 4);
  float am = fmaxf(fmaxf(fabsf(xv.x), fabsf(xv.y)), fmaxf(fabsf(xv.z), fabsf(xv.w)));
  am = warp_max32(am);
  const float amc   = fmaxf(am, 1e-12f);
  const float scale = amc * (1.f / 448.f);
  const float inv   = 448.f / amc;
  const float a = fminf(fmaxf(xv.x * inv, -448.f), 448.f);
  const float b = fminf(fmaxf(xv.y * inv, -448.f), 448.f);
  const float c = fminf(fmaxf(xv.z * inv, -448.f), 448.f);
  const float d = fminf(fmaxf(xv.w * inv, -448.f), 448.f);
  const unsigned pk = pack_fp8x4(a, b, c, d);
  const int c0 = kb * 128 + lane * 4;
#pragma unroll
  for (int j = 0; j < 4; ++j)
    XqT[(size_t)(c0 + j) * T_DIM + row] = (unsigned char)(pk >> (8 * j));
  if (lane == 0) Xs[row * NKB_H + kb] = scale;
}

__global__ __launch_bounds__(256) void moe_gemm1(
    const unsigned char* __restrict__ XqT, const float* __restrict__ Xs,
    const unsigned char* __restrict__ Wg,
    const float* __restrict__ Wgs,
    const int* __restrict__ tki,
    unsigned char* __restrict__ HqT, float* __restrict__ Hs, int e) {
  __shared__ __align__(16) unsigned char smem[2 * 40960 + 4096 + 1024 + 256 + 16];
  float* sxs    = (float*)(smem + 81920);
  float* part   = (float*)(smem + 86016);
  float* sc     = (float*)(smem + 87040);
  int*   routed = (int*)(smem + 87296);
  float* hb = (float*)smem;

  const int tid = threadIdx.x, lane = tid & 31, w = tid >> 5;
  const int wm = w >> 1, wn = w & 1;
  const int hi = (lane >> 4) << 3;
  const int m0 = blockIdx.x * 64;
  const int nb = blockIdx.y, n0 = nb * 128;

  if (tid == 0) *routed = 0;
  __syncthreads();
  if (tid < 128) {
    const int row = m0 + (tid >> 1);
    if (tki[row * 2 + (tid & 1)] == e) *routed = 1;
  }
  for (int i = tid; i < 64 * NKB_H; i += 256)
    sxs[i] = Xs[(m0 + (i >> 4)) * NKB_H + (i & 15)];
  __syncthreads();
  if (!*routed) return;

  float swg_a[NKB_H], swu_a[NKB_H];
#pragma unroll
  for (int kb = 0; kb < NKB_H; ++kb) {
    swg_a[kb] = Wgs[nb * NKB_H + kb];
    swu_a[kb] = Wgs[(NKB_I + nb) * NKB_H + kb];
  }

  v8f accg[4], accu[4];
#pragma unroll
  for (int t = 0; t < 4; ++t)
#pragma unroll
    for (int v = 0; v < 8; ++v) { accg[t][v] = 0.f; accu[t][v] = 0.f; }

  const unsigned b0 = lds_off_of(smem);

  auto stage = [&](unsigned base_o, int kb) {
#pragma unroll
    for (int c = 0; c < 4; ++c) {
      const int id = tid + c * 256;
      const int r = id >> 3, seg = (id & 7) * 16;
      async_copy_b128(Wg + (size_t)(n0 + r) * H_DIM + kb * 128 + seg,
                      base_o + r * 128 + seg);
      async_copy_b128(Wg + (size_t)(I_DIM + n0 + r) * H_DIM + kb * 128 + seg,
                      base_o + 16384 + r * 128 + seg);
    }
#pragma unroll
    for (int c = 0; c < 2; ++c) {
      const int id = tid + c * 256;
      const int r = id >> 2, seg = (id & 3) * 16;
      async_copy_b128(XqT + (size_t)(kb * 128 + r) * T_DIM + m0 + seg,
                      base_o + 32768 + r * 64 + seg);
    }
  };

  stage(b0, 0);
#pragma unroll 2
  for (int kb = 0; kb < NKB_H; ++kb) {
    if (kb + 1 < NKB_H) {
      stage(b0 + (unsigned)((kb + 1) & 1) * 40960u, kb + 1);
      wait_async_le10();
    } else {
      wait_async_le0();
    }
    __syncthreads();

    const unsigned char* wgl = smem + (kb & 1) * 40960;
    const unsigned char* wul = wgl + 16384;
    const unsigned char* xtl = wgl + 32768;

    const v16i Bx = lds_load_b_tok(xtl, lane, wm * 16);
    const float sxl = sxs[(wm * 16 + (lane & 15)) * NKB_H + kb];
    const float fg = sxl * swg_a[kb];
    const float fu = sxl * swu_a[kb];
#pragma unroll
    for (int t = 0; t < 4; ++t) {
      v8f zero = {0.f, 0.f, 0.f, 0.f, 0.f, 0.f, 0.f, 0.f};
      const v16i Ag = lds_load_a(wgl + (wn * 64 + t * 16) * 128, lane);
      const v8f pg = __builtin_amdgcn_wmma_f32_16x16x128_fp8_fp8(Ag, Bx, (short)0, zero, false, false);
      const v16i Au = lds_load_a(wul + (wn * 64 + t * 16) * 128, lane);
      const v8f pu = __builtin_amdgcn_wmma_f32_16x16x128_fp8_fp8(Au, Bx, (short)0, zero, false, false);
#pragma unroll
      for (int v = 0; v < 8; ++v) {
        accg[t][v] += pg[v] * fg;
        accu[t][v] += pu[v] * fu;
      }
    }
    __syncthreads();
  }

#pragma unroll
  for (int t = 0; t < 4; ++t)
#pragma unroll
    for (int v = 0; v < 8; ++v) {
      const float g = accg[t][v];
      const float h = (g / (1.f + __expf(-g))) * accu[t][v];
      const int nl = wn * 64 + t * 16 + v + hi;
      const int ml = wm * 16 + (lane & 15);
      hb[nl * 68 + ml] = h;
    }
  __syncthreads();

  {
    const int tok = tid & 63, qtr = tid >> 6;
    float am = 0.f;
    for (int n = qtr * 32; n < qtr * 32 + 32; ++n)
      am = fmaxf(am, fabsf(hb[n * 68 + tok]));
    part[qtr * 64 + tok] = am;
  }
  __syncthreads();
  if (tid < 64) {
    const float am = fmaxf(fmaxf(part[tid], part[64 + tid]),
                           fmaxf(part[128 + tid], part[192 + tid]));
    const float amc = fmaxf(am, 1e-12f);
    sc[tid] = 448.f / amc;
    Hs[(m0 + tid) * NKB_H + nb] = amc * (1.f / 448.f);
  }
  __syncthreads();

#pragma unroll
  for (int ii = 0; ii < 8; ++ii) {
    const int id = ii * 256 + tid;
    const int n = id >> 4, mg = (id & 15) * 4;
    float q[4];
#pragma unroll
    for (int j = 0; j < 4; ++j)
      q[j] = fminf(fmaxf(hb[n * 68 + mg + j] * sc[mg + j], -448.f), 448.f);
    *reinterpret_cast<unsigned*>(HqT + (size_t)(n0 + n) * T_DIM + m0 + mg) =
        pack_fp8x4(q[0], q[1], q[2], q[3]);
  }
}

__global__ __launch_bounds__(256) void moe_gemm2(
    const unsigned char* __restrict__ HqT, const float* __restrict__ Hs,
    const unsigned char* __restrict__ Wd,
    const float* __restrict__ Wds,
    const int* __restrict__ tki, const float* __restrict__ tkw,
    float* __restrict__ out, int e) {
  __shared__ __align__(16) unsigned char smem[2 * 24576 + 4096 + 256 + 16];
  float* shs    = (float*)(smem + 49152);
  float* rwb    = (float*)(smem + 53248);
  int*   routed = (int*)(smem + 53504);

  const int tid = threadIdx.x, lane = tid & 31, w = tid >> 5;
  const int wm = w >> 1, wn = w & 1;
  const int hi = (lane >> 4) << 3;
  const int m0 = blockIdx.x * 64;
  const int nb = blockIdx.y, n0 = nb * 128;

  if (tid == 0) *routed = 0;
  __syncthreads();
  if (tid < 64) {
    const int row = m0 + tid;
    float rw = 0.f;
#pragma unroll
    for (int j = 0; j < 2; ++j)
      if (tki[row * 2 + j] == e) rw += tkw[row * 2 + j];
    rwb[tid] = rw;
    if (rw != 0.f) *routed = 1;
  }
  for (int i = tid; i < 64 * NKB_H; i += 256)
    shs[i] = Hs[(m0 + (i >> 4)) * NKB_H + (i & 15)];
  __syncthreads();
  if (!*routed) return;

  float swd_a[NKB_I];
#pragma unroll
  for (int kb = 0; kb < NKB_I; ++kb) swd_a[kb] = Wds[nb * NKB_I + kb];

  v8f acc[4];
#pragma unroll
  for (int t = 0; t < 4; ++t)
#pragma unroll
    for (int v = 0; v < 8; ++v) acc[t][v] = 0.f;

  const unsigned b0 = lds_off_of(smem);

  auto stage = [&](unsigned base_o, int kb) {
#pragma unroll
    for (int c = 0; c < 4; ++c) {
      const int id = tid + c * 256;
      const int r = id >> 3, seg = (id & 7) * 16;
      async_copy_b128(Wd + (size_t)(n0 + r) * I_DIM + kb * 128 + seg,
                      base_o + r * 128 + seg);
    }
#pragma unroll
    for (int c = 0; c < 2; ++c) {
      const int id = tid + c * 256;
      const int r = id >> 2, seg = (id & 3) * 16;
      async_copy_b128(HqT + (size_t)(kb * 128 + r) * T_DIM + m0 + seg,
                      base_o + 16384 + r * 64 + seg);
    }
  };

  stage(b0, 0);
#pragma unroll 2
  for (int kb = 0; kb < NKB_I; ++kb) {
    if (kb + 1 < NKB_I) {
      stage(b0 + (unsigned)((kb + 1) & 1) * 24576u, kb + 1);
      wait_async_le6();
    } else {
      wait_async_le0();
    }
    __syncthreads();

    const unsigned char* wdl = smem + (kb & 1) * 24576;
    const unsigned char* htl = wdl + 16384;

    const v16i Bx = lds_load_b_tok(htl, lane, wm * 16);
    const float f = shs[(wm * 16 + (lane & 15)) * NKB_H + kb] * swd_a[kb];
#pragma unroll
    for (int t = 0; t < 4; ++t) {
      v8f zero = {0.f, 0.f, 0.f, 0.f, 0.f, 0.f, 0.f, 0.f};
      const v16i Ad = lds_load_a(wdl + (wn * 64 + t * 16) * 128, lane);
      const v8f p = __builtin_amdgcn_wmma_f32_16x16x128_fp8_fp8(Ad, Bx, (short)0, zero, false, false);
#pragma unroll
      for (int v = 0; v < 8; ++v) acc[t][v] += p[v] * f;
    }
    __syncthreads();
  }

#pragma unroll
  for (int t = 0; t < 4; ++t)
#pragma unroll
    for (int v = 0; v < 8; ++v) {
      const int nl = wn * 64 + t * 16 + v + hi;
      const int ml = wm * 16 + (lane & 15);
      float* o = out + (size_t)(m0 + ml) * H_DIM + n0 + nl;
      *o += rwb[ml] * acc[t][v];
    }
}


extern "C" void kernel_launch(void* const* d_in, const int* in_sizes, int n_in,
                              void* d_out, int out_size, void* d_ws, size_t ws_size,
                              hipStream_t stream) {
  const float*         X    = (const float*)d_in[0];
  const int*           tki  = (const int*)d_in[1];
  const float*         tkw  = (const float*)d_in[2];
  const unsigned char* gup  = (const unsigned char*)d_in[3];
  const float*         gups = (const float*)d_in[4];
  const unsigned char* dwn  = (const unsigned char*)d_in[5];
  const float*         dwns = (const float*)d_in[6];
  float*               out  = (float*)d_out;

  char* ws = (char*)d_ws;
  unsigned char* XqT = (unsigned char*)ws;
  float*         Xs  = (float*)(ws + (4u << 20));
  unsigned char* HqT = (unsigned char*)(ws + (4u << 20) + (256u << 10));
  float*         Hs  = (float*)(ws + (7u << 20) + (256u << 10));

  moe_zero<<<(T_DIM * H_DIM + 255) / 256, 256, 0, stream>>>(out, T_DIM * H_DIM);
  moe_quant_x<<<(T_DIM * NKB_H) / 8, 256, 0, stream>>>(X, XqT, Xs);

  for (int e = 0; e < 8; ++e) {
    moe_gemm1<<<dim3(T_DIM / 64, NKB_I), 256, 0, stream>>>(
        XqT, Xs, gup + (size_t)e * (2 * I_DIM) * H_DIM, gups + e * 22 * NKB_H,
        tki, HqT, Hs, e);
    moe_gemm2<<<dim3(T_DIM / 64, H_DIM / 128), 256, 0, stream>>>(
        HqT, Hs, dwn + (size_t)e * H_DIM * I_DIM, dwns + e * 16 * NKB_I,
        tki, tkw, out, e);
  }
}
